// BaselineNER_35476429865753
// MI455X (gfx1250) — matrix operand order repaired and checked by reading, not compiled, not run on hardware
//
#include <hip/hip_runtime.h>
#include <cstdint>
#include <cstddef>

// ---------------------------------------------------------------------------
// BiLSTM-CRF forward for MI455X (gfx1250, wave32, WMMA f16->f32 accumulate).
// Software-pipelined K loops: next slab's fragments are loaded while the
// current slab's v_wmma instructions execute.
// ---------------------------------------------------------------------------

typedef __attribute__((ext_vector_type(16))) _Float16 v16h;
typedef __attribute__((ext_vector_type(8)))  _Float16 v8h;
typedef __attribute__((ext_vector_type(8)))  float    v8f;

#define BATCH 64
#define SEQ   512
#define EMB   300
#define EMBP  320      // K padded to multiple of 32 for 16x16x32 WMMA
#define HID   256
#define G4    1024     // 4*HID gates
#define NTAG  25
#define NTOK  (BATCH*SEQ)

// ---------------- WMMA fragment loaders (wave32) ----------------
// A 16x32 f16 as loaded here (contiguous k, same order as ld_frag_bt): lanes 0-15 hold row M=lane with
// K{0..15}; lanes 16-31 hold row M=lane-16 with K{16..31}.
__device__ __forceinline__ v16h ld_frag_a(const _Float16* A, int lda, int row0, int k0) {
  const int lane = threadIdx.x & 31;
  const _Float16* p = A + (size_t)(row0 + (lane & 15)) * lda + (k0 + ((lane >> 4) << 4));
  v16h f;
  v8h* fp = (v8h*)&f;
  fp[0] = *(const v8h*)p;        // 16B contiguous
  fp[1] = *(const v8h*)(p + 8);  // next 16B, K+8
  return f;
}
// B-matrix 32x16 f16 where B = W^T and W is row-major [N][K]:
// lanes 0-15: column N=lane, K k0..k0+15; lanes 16-31: N=lane-16, K k0+16..k0+31.
__device__ __forceinline__ v16h ld_frag_bt(const _Float16* W, int ldw, int n0, int k0) {
  const int lane = threadIdx.x & 31;
  const _Float16* p = W + (size_t)(n0 + (lane & 15)) * ldw + (k0 + ((lane >> 4) << 4));
  return *(const v16h*)p;        // 32B contiguous
}

__device__ __forceinline__ float sigf(float x)   { return 1.0f / (1.0f + __expf(-x)); }
__device__ __forceinline__ float tanhf_(float x) { return 1.0f - 2.0f / (__expf(2.0f * x) + 1.0f); }

__device__ __forceinline__ v8f wmma_f16(v16h a, v16h b, v8f c) {
  return __builtin_amdgcn_wmma_f32_16x16x32_f16(false, a, false, b, (short)0, c, false, false);
}

// ---------------- weight convert f32 -> f16 with K padding ----------------
__global__ void k_cvt(const float* __restrict__ src, _Float16* __restrict__ dst,
                      int rows, int K, int Kp) {
  int i = blockIdx.x * blockDim.x + threadIdx.x;
  if (i >= rows * Kp) return;
  int r = i / Kp, k = i % Kp;
  dst[i] = (k < K) ? (_Float16)src[r * K + k] : (_Float16)0.0f;
}

// ---------------- embedding gather + convert ----------------
__global__ void k_embed(const int* __restrict__ sent, const float* __restrict__ embed,
                        _Float16* __restrict__ x16) {
  size_t i = (size_t)blockIdx.x * blockDim.x + threadIdx.x;
  if (i >= (size_t)NTOK * EMBP) return;
  int m = (int)(i / EMBP);
  int e = (int)(i % EMBP);
  _Float16 v = (_Float16)0.0f;
  if (e < EMB) v = (_Float16)embed[(size_t)sent[m] * EMB + e];
  x16[i] = v;
}

// ---------------- input projection GEMM: pre = x @ Wih^T + b ----------------
// grid (2048, 16, 2): x = M-tile of 16 tokens, y = 4 N-tiles of 16, z = direction.
// One wave per block; stores pre time-major [T][B][1024] for the recurrence.
// K loop is software pipelined (next slab loaded during current WMMAs).
__global__ void k_gemm_pre(const _Float16* __restrict__ x16,
                           const _Float16* __restrict__ wf, const _Float16* __restrict__ wb,
                           const float* __restrict__ biasf, const float* __restrict__ biasb,
                           float* __restrict__ pref, float* __restrict__ preb) {
  const int dir  = blockIdx.z;
  const int m0   = blockIdx.x * 16;
  const int n0   = blockIdx.y * 64;
  const int lane = threadIdx.x & 31;
  const _Float16* W    = dir ? wb : wf;
  const float*    bias = dir ? biasb : biasf;
  float*          pre  = dir ? preb : pref;

  v8f acc0 = {}, acc1 = {}, acc2 = {}, acc3 = {};

  // prologue: slab 0 fragments
  v16h a  = ld_frag_a(x16, EMBP, m0, 0);
  v16h b0 = ld_frag_bt(W, EMBP, n0 +  0, 0);
  v16h b1 = ld_frag_bt(W, EMBP, n0 + 16, 0);
  v16h b2 = ld_frag_bt(W, EMBP, n0 + 32, 0);
  v16h b3 = ld_frag_bt(W, EMBP, n0 + 48, 0);

  const _Float16* arow = x16 + (size_t)(m0 + (lane & 15)) * EMBP;
  #pragma unroll
  for (int kb = 0; kb < EMBP; kb += 32) {
    const int kn = kb + 32;
    v16h an = a, c0 = b0, c1 = b1, c2 = b2, c3 = b3;
    if (kn < EMBP) {
      __builtin_prefetch(arow + kn, 0, 3);
      an = ld_frag_a(x16, EMBP, m0, kn);
      c0 = ld_frag_bt(W, EMBP, n0 +  0, kn);
      c1 = ld_frag_bt(W, EMBP, n0 + 16, kn);
      c2 = ld_frag_bt(W, EMBP, n0 + 32, kn);
      c3 = ld_frag_bt(W, EMBP, n0 + 48, kn);
    }
    acc0 = wmma_f16(a, b0, acc0);
    acc1 = wmma_f16(a, b1, acc1);
    acc2 = wmma_f16(a, b2, acc2);
    acc3 = wmma_f16(a, b3, acc3);
    a = an; b0 = c0; b1 = c1; b2 = c2; b3 = c3;
  }

  // D layout: VGPR r -> M = r (lanes 0-15) / r+8 (lanes 16-31); N = lane&15.
  const int mh = (lane >> 4) << 3;
  #pragma unroll
  for (int nt = 0; nt < 4; ++nt) {
    v8f acc = (nt == 0) ? acc0 : (nt == 1) ? acc1 : (nt == 2) ? acc2 : acc3;
    int n = n0 + nt * 16 + (lane & 15);
    float bv = bias[n];
    #pragma unroll
    for (int r = 0; r < 8; ++r) {
      int m = m0 + r + mh;           // token index = b*SEQ + t
      int bb = m >> 9, tt = m & (SEQ - 1);
      pre[((size_t)tt * BATCH + bb) * G4 + n] = acc[r] + bv;
    }
  }
}

// ---------------- zero h/c state ----------------
__global__ void k_zero(_Float16* __restrict__ h16, float* __restrict__ c) {
  int i = blockIdx.x * blockDim.x + threadIdx.x;
  if (i < 2 * 2 * BATCH * HID) h16[i] = (_Float16)0.0f;
  if (i < 2 * BATCH * HID)     c[i]   = 0.0f;
}

// ---------------- one LSTM time step, both directions ----------------
// grid (4, 16, 2): x = batch tile, y = hidden tile j0, z = direction. 1 wave.
// Each wave computes the 4 gate tiles {j, H+j, 2H+j, 3H+j} (32 WMMAs, pipelined)
// and fuses the elementwise cell update.
__global__ void k_step(int s,
                       const _Float16* __restrict__ whhf, const _Float16* __restrict__ whhb,
                       const float* __restrict__ pref, const float* __restrict__ preb,
                       _Float16* __restrict__ h16, float* __restrict__ cst,
                       float* __restrict__ houtf, float* __restrict__ houtb) {
  const int dir  = blockIdx.z;
  const int t    = dir ? (SEQ - 1 - s) : s;
  const int b0   = blockIdx.x * 16;
  const int j0   = blockIdx.y * 16;
  const int lane = threadIdx.x & 31;

  const _Float16* W     = dir ? whhb : whhf;
  const float*    pre   = (dir ? preb : pref) + (size_t)t * BATCH * G4;
  const _Float16* hprev = h16 + ((size_t)dir * 2 + (s & 1)) * BATCH * HID;
  _Float16*       hnext = h16 + ((size_t)dir * 2 + ((s + 1) & 1)) * BATCH * HID;
  float*          c     = cst + (size_t)dir * BATCH * HID;
  float*          hout  = dir ? houtb : houtf;

  v8f ai = {}, af = {}, ag = {}, ao = {};

  v16h a  = ld_frag_a(hprev, HID, b0, 0);
  v16h bi = ld_frag_bt(W, HID,           j0, 0);
  v16h bf = ld_frag_bt(W, HID,   HID +   j0, 0);
  v16h bg = ld_frag_bt(W, HID, 2*HID +   j0, 0);
  v16h bo = ld_frag_bt(W, HID, 3*HID +   j0, 0);

  #pragma unroll
  for (int kb = 0; kb < HID; kb += 32) {
    const int kn = kb + 32;
    v16h an = a, ci = bi, cf = bf, cg = bg, co = bo;
    if (kn < HID) {
      an = ld_frag_a(hprev, HID, b0, kn);
      ci = ld_frag_bt(W, HID,           j0, kn);
      cf = ld_frag_bt(W, HID,   HID +   j0, kn);
      cg = ld_frag_bt(W, HID, 2*HID +   j0, kn);
      co = ld_frag_bt(W, HID, 3*HID +   j0, kn);
    }
    ai = wmma_f16(a, bi, ai);
    af = wmma_f16(a, bf, af);
    ag = wmma_f16(a, bg, ag);
    ao = wmma_f16(a, bo, ao);
    a = an; bi = ci; bf = cf; bg = cg; bo = co;
  }

  const int j  = j0 + (lane & 15);
  const int mh = (lane >> 4) << 3;
  #pragma unroll
  for (int r = 0; r < 8; ++r) {
    int m = b0 + r + mh;                       // batch index
    const float* pm = pre + (size_t)m * G4;
    float gi = ai[r] + pm[j];
    float gf = af[r] + pm[HID + j];
    float gg = ag[r] + pm[2 * HID + j];
    float go = ao[r] + pm[3 * HID + j];
    size_t cix = (size_t)m * HID + j;
    float cn = sigf(gf) * c[cix] + sigf(gi) * tanhf_(gg);
    float h  = sigf(go) * tanhf_(cn);
    c[cix] = cn;
    hnext[cix] = (_Float16)h;
    hout[((size_t)m * SEQ + t) * HID + j] = h;  // [B][T][H]
  }
}

// ---------------- emissions: em = [h_f, h_b] @ Wlin^T + blin ----------------
__global__ void k_emis(const float* __restrict__ hf, const float* __restrict__ hb,
                       const float* __restrict__ Wlin, const float* __restrict__ blin,
                       float* __restrict__ em) {
  int tok  = blockIdx.x * 8 + (threadIdx.x >> 5);
  int k    = threadIdx.x & 31;
  if (k >= NTAG) return;
  const float* xf = hf + (size_t)tok * HID;
  const float* xb = hb + (size_t)tok * HID;
  const float* w  = Wlin + (size_t)k * (2 * HID);
  float s = blin[k];
  #pragma unroll 4
  for (int jj = 0; jj < HID; ++jj) s += xf[jj] * w[jj] + xb[jj] * w[HID + jj];
  em[(size_t)tok * NTAG + k] = s;
}

// ---------------- CRF log-likelihood per batch row (one wave each) ----------
__global__ void k_crf(const float* __restrict__ em, const int* __restrict__ tags,
                      const float* __restrict__ start, const float* __restrict__ end,
                      const float* __restrict__ trans, float* __restrict__ part) {
  const int b = blockIdx.x;
  const int lane = threadIdx.x;
  __shared__ float str[NTAG * NTAG];
  for (int i = lane; i < NTAG * NTAG; i += 32) str[i] = trans[i];
  __syncthreads();
  const int*   tg = tags + b * SEQ;
  const float* e  = em + (size_t)b * SEQ * NTAG;

  // numerator: emission + transition scores along the gold path
  float nsum = 0.0f;
  for (int t = lane; t < SEQ; t += 32) {
    int cur = tg[t];
    nsum += e[(size_t)t * NTAG + cur];
    if (t > 0) nsum += str[tg[t - 1] * NTAG + cur];
  }
  for (int off = 16; off; off >>= 1) nsum += __shfl_xor(nsum, off, 32);

  // forward algorithm: lane j holds alpha[j]; trans column preloaded to regs
  int jj = (lane < NTAG) ? lane : 0;
  float colj[NTAG];
  #pragma unroll
  for (int k = 0; k < NTAG; ++k) colj[k] = str[k * NTAG + jj];

  float alpha = (lane < NTAG) ? (start[jj] + e[jj]) : -3.0e38f;
  for (int t = 1; t < SEQ; ++t) {
    float M = -3.0e38f, S = 0.0f;          // online logsumexp over prev states
    #pragma unroll
    for (int k = 0; k < NTAG; ++k) {
      float a  = __shfl(alpha, k, 32) + colj[k];
      float nm = fmaxf(M, a);
      S = S * __expf(M - nm) + __expf(a - nm);
      M = nm;
    }
    float ee = (lane < NTAG) ? e[(size_t)t * NTAG + jj] : 0.0f;
    alpha = M + __logf(S) + ee;
  }
  float v  = (lane < NTAG) ? (alpha + end[jj]) : -3.0e38f;
  float mx = v;
  for (int off = 16; off; off >>= 1) mx = fmaxf(mx, __shfl_xor(mx, off, 32));
  float sm = (lane < NTAG) ? __expf(v - mx) : 0.0f;
  for (int off = 16; off; off >>= 1) sm += __shfl_xor(sm, off, 32);
  float denom = mx + __logf(sm);

  if (lane == 0) {
    float num = nsum + start[tg[0]] + end[tg[SEQ - 1]];
    part[b] = num - denom;
  }
}

// deterministic ordered final reduction (no float atomics)
__global__ void k_final(const float* __restrict__ part, float* __restrict__ out) {
  if (threadIdx.x == 0 && blockIdx.x == 0) {
    float s = 0.0f;
    for (int i = 0; i < BATCH; ++i) s += part[i];
    out[0] = s;
  }
}

// ---------------- workspace layout (bytes, 256-aligned) ----------------
static constexpr size_t SZ_WIH16 = (size_t)G4 * EMBP * 2;       // 655,360
static constexpr size_t SZ_WHH16 = (size_t)G4 * HID * 2;        // 524,288
static constexpr size_t SZ_X16   = (size_t)NTOK * EMBP * 2;     // 20,971,520
static constexpr size_t SZ_PRE   = (size_t)SEQ * BATCH * G4 * 4;// 134,217,728
static constexpr size_t SZ_H     = (size_t)BATCH * SEQ * HID * 4;// 33,554,432
static constexpr size_t SZ_H16   = (size_t)2 * 2 * BATCH * HID * 2;
static constexpr size_t SZ_C     = (size_t)2 * BATCH * HID * 4;
static constexpr size_t SZ_EM    = (size_t)NTOK * NTAG * 4;

static constexpr size_t OFF_WIH16F = 0;
static constexpr size_t OFF_WIH16B = OFF_WIH16F + SZ_WIH16;
static constexpr size_t OFF_WHH16F = OFF_WIH16B + SZ_WIH16;
static constexpr size_t OFF_WHH16B = OFF_WHH16F + SZ_WHH16;
static constexpr size_t OFF_X16    = OFF_WHH16B + SZ_WHH16;
static constexpr size_t OFF_PREF   = OFF_X16 + SZ_X16;
static constexpr size_t OFF_PREB   = OFF_PREF + SZ_PRE;
static constexpr size_t OFF_HF     = OFF_PREB + SZ_PRE;
static constexpr size_t OFF_HB     = OFF_HF + SZ_H;
static constexpr size_t OFF_H16    = OFF_HB + SZ_H;
static constexpr size_t OFF_C      = OFF_H16 + SZ_H16;
static constexpr size_t OFF_EM     = OFF_C + SZ_C;
static constexpr size_t OFF_PART   = OFF_EM + SZ_EM;
// total ~363 MB of d_ws used

extern "C" void kernel_launch(void* const* d_in, const int* in_sizes, int n_in,
                              void* d_out, int out_size, void* d_ws, size_t ws_size,
                              hipStream_t stream) {
  (void)in_sizes; (void)n_in; (void)out_size; (void)ws_size;
  const int*   sent   = (const int*)d_in[0];
  const int*   tags   = (const int*)d_in[1];
  const float* embed  = (const float*)d_in[2];
  const float* Wih_f  = (const float*)d_in[3];
  const float* Whh_f  = (const float*)d_in[4];
  const float* b_f    = (const float*)d_in[5];
  const float* Wih_b  = (const float*)d_in[6];
  const float* Whh_b  = (const float*)d_in[7];
  const float* b_b    = (const float*)d_in[8];
  const float* Wlin   = (const float*)d_in[9];
  const float* blin   = (const float*)d_in[10];
  const float* cstart = (const float*)d_in[11];
  const float* cend   = (const float*)d_in[12];
  const float* ctrans = (const float*)d_in[13];

  char* ws = (char*)d_ws;
  _Float16* wih16f = (_Float16*)(ws + OFF_WIH16F);
  _Float16* wih16b = (_Float16*)(ws + OFF_WIH16B);
  _Float16* whh16f = (_Float16*)(ws + OFF_WHH16F);
  _Float16* whh16b = (_Float16*)(ws + OFF_WHH16B);
  _Float16* x16    = (_Float16*)(ws + OFF_X16);
  float*    pref   = (float*)(ws + OFF_PREF);
  float*    preb   = (float*)(ws + OFF_PREB);
  float*    houtf  = (float*)(ws + OFF_HF);
  float*    houtb  = (float*)(ws + OFF_HB);
  _Float16* h16    = (_Float16*)(ws + OFF_H16);
  float*    cbuf   = (float*)(ws + OFF_C);
  float*    em     = (float*)(ws + OFF_EM);
  float*    part   = (float*)(ws + OFF_PART);

  // 1. weight conversion to f16 (K-padded)
  k_cvt<<<(G4 * EMBP + 255) / 256, 256, 0, stream>>>(Wih_f, wih16f, G4, EMB, EMBP);
  k_cvt<<<(G4 * EMBP + 255) / 256, 256, 0, stream>>>(Wih_b, wih16b, G4, EMB, EMBP);
  k_cvt<<<(G4 * HID  + 255) / 256, 256, 0, stream>>>(Whh_f, whh16f, G4, HID, HID);
  k_cvt<<<(G4 * HID  + 255) / 256, 256, 0, stream>>>(Whh_b, whh16b, G4, HID, HID);

  // 2. embedding gather + f16 convert
  k_embed<<<(unsigned)(((size_t)NTOK * EMBP + 255) / 256), 256, 0, stream>>>(sent, embed, x16);

  // 3. batched input projection for both directions (WMMA, pipelined)
  {
    dim3 g(NTOK / 16, 16, 2);
    k_gemm_pre<<<g, 32, 0, stream>>>(x16, wih16f, wih16b, b_f, b_b, pref, preb);
  }

  // 4. recurrence: zero state, then 512 sequential steps (both dirs per launch)
  k_zero<<<(2 * 2 * BATCH * HID + 255) / 256, 256, 0, stream>>>(h16, cbuf);
  for (int s = 0; s < SEQ; ++s) {
    dim3 g(BATCH / 16, HID / 16, 2);
    k_step<<<g, 32, 0, stream>>>(s, whh16f, whh16b, pref, preb, h16, cbuf, houtf, houtb);
  }

  // 5. emissions
  k_emis<<<NTOK / 8, 256, 0, stream>>>(houtf, houtb, Wlin, blin, em);

  // 6. CRF log-likelihood, deterministic ordered sum
  k_crf<<<BATCH, 32, 0, stream>>>(em, tags, cstart, cend, ctrans, part);
  k_final<<<1, 32, 0, stream>>>(part, (float*)d_out);
}
